// MambaBlockTPU_43860206027426
// MI455X (gfx1250) — matrix operand order repaired and checked by reading, not compiled, not run on hardware
//
#include <hip/hip_runtime.h>
#include <hip/hip_bf16.h>
#include <stdint.h>

#define B_  2
#define L_  1024
#define DIN 1024
#define DM  2048
#define DS  16
#define DR  128
#define KC  4
#define T_  (B_ * L_)

typedef __attribute__((ext_vector_type(16))) __bf16 v16bf;
typedef __attribute__((ext_vector_type(8)))  float  v8f;
typedef __attribute__((ext_vector_type(4)))  unsigned int v4u;

__device__ __forceinline__ unsigned short f2bf(float f) {
    unsigned int u = __float_as_uint(f);
    u += 0x7fffu + ((u >> 16) & 1u);          // round-to-nearest-even
    return (unsigned short)(u >> 16);
}

// ---------------- elementwise cast: f32 -> bf16 (row major) ----------------
__global__ void __launch_bounds__(256)
cast_bf16_kernel(const float* __restrict__ src, unsigned short* __restrict__ dst, int n) {
    int i = blockIdx.x * 256 + threadIdx.x;
    if (i < n) dst[i] = f2bf(src[i]);
}

// -------- pack weight (Krows x Ncols, row major f32) -> bf16 (Ncols x Krows) --------
__global__ void __launch_bounds__(256)
pack_transpose_kernel(const float* __restrict__ src, unsigned short* __restrict__ dst,
                      int Krows, int Ncols) {
    int i = blockIdx.x * 256 + threadIdx.x;
    if (i >= Krows * Ncols) return;
    int n = i / Krows;
    int k = i % Krows;
    dst[i] = f2bf(src[(size_t)k * Ncols + n]);
}

// ===================== LDS-tiled bf16 WMMA GEMM =====================
// C(MxN,f32) = A(MxK,bf16 row-major) * BT(NxK,bf16)^T
// Block: 256 threads = 8 waves as 2(M) x 4(N).
// Wave tile: 32 x (16*TN)  => block tile 64 x (64*TN).  K chunked by 32 via LDS.
// Requires M%64==0, N%(64*TN)==0, K%32==0.
// Fragment layouts per CDNA5 ISA 7.12.2 (wave32):
//   A 16x32 bf16 : lane m=l%16; e<8 -> K = 8*(l>=16)+e ; e>=8 -> K = 16+8*(l>=16)+(e-8)
//   B 32x16 bf16 : lane n=l%16; e<8 -> K = 8*(l>=16)+e ; e>=8 -> K = 16+8*(l>=16)+(e-8)
//   C/D 16x16 f32: vgpr r -> M = r + 8*(l>=16), N = l%16
template <int TN>
__global__ void __launch_bounds__(256)
gemm_bf16_wmma_tiled(const unsigned short* __restrict__ A,
                     const unsigned short* __restrict__ BT,
                     float* __restrict__ C,
                     unsigned short* __restrict__ Cb,   // optional bf16 dual output
                     int M, int N, int Kd) {
    __shared__ __align__(16) unsigned short lA[64 * 32];            // 4 KB
    __shared__ __align__(16) unsigned short lB[64 * TN * 32];       // 4*TN KB

    const int tid  = threadIdx.x;
    const int lane = tid & 31;
    const int wv   = tid >> 5;
    const int wm   = wv >> 2;        // 0..1  (M dir)
    const int wn   = wv & 3;         // 0..3  (N dir)
    const int l15  = lane & 15;
    const int hi   = lane >> 4;
    const int kbA  = hi * 8;
    const int kbB  = hi * 8;

    const int blockM = blockIdx.y * 64;
    const int blockN = blockIdx.x * (64 * TN);

    // staging coords (constant across iterations)
    const int sRow = tid >> 2;          // 0..63
    const int sKK  = (tid & 3) * 8;     // 0,8,16,24

    union Frag { v4u u[2]; v16bf v; };

    v8f acc[2][TN];
#pragma unroll
    for (int i = 0; i < 2; ++i)
#pragma unroll
        for (int j = 0; j < TN; ++j) acc[i][j] = (v8f){};

    for (int k = 0; k < Kd; k += 32) {
        // ---- stage A (64x32) and B ((64*TN)x32) into LDS ----
        {
            v4u av = *reinterpret_cast<const v4u*>(A + (size_t)(blockM + sRow) * Kd + k + sKK);
            *reinterpret_cast<v4u*>(lA + sRow * 32 + sKK) = av;
#pragma unroll
            for (int j = 0; j < TN; ++j) {
                v4u bv = *reinterpret_cast<const v4u*>(
                    BT + (size_t)(blockN + j * 64 + sRow) * Kd + k + sKK);
                *reinterpret_cast<v4u*>(lB + (j * 64 + sRow) * 32 + sKK) = bv;
            }
        }
        __syncthreads();

        // ---- load fragments from LDS ----
        Frag fa[2], fb[TN];
#pragma unroll
        for (int tm2 = 0; tm2 < 2; ++tm2) {
            int row = wm * 32 + tm2 * 16 + l15;
            fa[tm2].u[0] = *reinterpret_cast<const v4u*>(lA + row * 32 + kbA);
            fa[tm2].u[1] = *reinterpret_cast<const v4u*>(lA + row * 32 + 16 + kbA);
        }
#pragma unroll
        for (int tn2 = 0; tn2 < TN; ++tn2) {
            int n = wn * (16 * TN) + tn2 * 16 + l15;
            fb[tn2].u[0] = *reinterpret_cast<const v4u*>(lB + n * 32 + kbB);
            fb[tn2].u[1] = *reinterpret_cast<const v4u*>(lB + n * 32 + kbB + 16);
        }

        // ---- 2 x TN wmma ----
#pragma unroll
        for (int tm2 = 0; tm2 < 2; ++tm2)
#pragma unroll
            for (int tn2 = 0; tn2 < TN; ++tn2)
                acc[tm2][tn2] = __builtin_amdgcn_wmma_f32_16x16x32_bf16(
                    false, fa[tm2].v, false, fb[tn2].v, (short)0, acc[tm2][tn2], false, false);

        __syncthreads();
    }

    // ---- epilogue ----
#pragma unroll
    for (int tm2 = 0; tm2 < 2; ++tm2) {
        const int crow0 = blockM + wm * 32 + tm2 * 16 + hi * 8;
#pragma unroll
        for (int tn2 = 0; tn2 < TN; ++tn2) {
            const int ccol = blockN + wn * (16 * TN) + tn2 * 16 + l15;
#pragma unroll
            for (int r = 0; r < 8; ++r) {
                size_t ci = (size_t)(crow0 + r) * N + ccol;
                C[ci] = acc[tm2][tn2][r];
                if (Cb) Cb[ci] = f2bf(acc[tm2][tn2][r]);
            }
        }
    }
}

// ===================== simple per-wave WMMA GEMM (small N, e.g. N=16) =====================
__global__ void __launch_bounds__(256)
gemm_bf16_wmma(const unsigned short* __restrict__ A,
               const unsigned short* __restrict__ BT,
               float* __restrict__ C,
               unsigned short* __restrict__ Cb,
               int M, int N, int Kd) {
    const int lane = threadIdx.x & 31;
    const int wv   = threadIdx.x >> 5;
    const int tilesN = N >> 4;
    const int nTiles = (M >> 4) * tilesN;
    const int tile = blockIdx.x * 8 + wv;
    if (tile >= nTiles) return;
    const int tm  = tile / tilesN;
    const int tn  = tile % tilesN;
    const int l15 = lane & 15;
    const int hi  = lane >> 4;

    const size_t arow = (size_t)(tm * 16 + l15) * Kd;
    const size_t brow = (size_t)(tn * 16 + l15) * Kd;
    const int kbA = hi * 8;
    const int kbB = hi * 8;

    union Frag { v4u u[2]; v16bf v; };
    v8f acc = {};
    for (int k = 0; k < Kd; k += 32) {
        Frag fa, fb;
        fa.u[0] = *reinterpret_cast<const v4u*>(A + arow + k + kbA);
        fa.u[1] = *reinterpret_cast<const v4u*>(A + arow + k + 16 + kbA);
        fb.u[0] = *reinterpret_cast<const v4u*>(BT + brow + k + kbB);
        fb.u[1] = *reinterpret_cast<const v4u*>(BT + brow + k + kbB + 16);
        acc = __builtin_amdgcn_wmma_f32_16x16x32_bf16(
                  false, fa.v, false, fb.v, (short)0, acc, false, false);
    }

    const int ccol  = tn * 16 + l15;
    const int crow0 = tm * 16 + hi * 8;
#pragma unroll
    for (int r = 0; r < 8; ++r) {
        size_t ci = (size_t)(crow0 + r) * N + ccol;
        C[ci] = acc[r];
        if (Cb) Cb[ci] = f2bf(acc[r]);
    }
}

// ---------------- causal depthwise conv (K=4) + SiLU; emits f32 + bf16 ----------------
__global__ void __launch_bounds__(256)
conv_silu_kernel(const float* __restrict__ x, const float* __restrict__ cw,
                 const float* __restrict__ cb, float* __restrict__ xc,
                 unsigned short* __restrict__ xcb) {
    int idx = blockIdx.x * 256 + threadIdx.x;
    if (idx >= T_ * DM) return;
    int d  = idx % DM;
    int tl = idx / DM;
    int t  = tl % L_;                       // position within sequence (pad per-sequence)
    float acc = cb[d];
#pragma unroll
    for (int i = 0; i < KC; ++i) {
        int ts = t + i - (KC - 1);
        if (ts >= 0) acc += x[idx + (i - (KC - 1)) * DM] * cw[d * KC + i];
    }
    float s = acc * (1.0f / (1.0f + __expf(-acc)));
    xc[idx]  = s;
    xcb[idx] = f2bf(s);
}

// ---------------- softplus(z + dt_b) in place ----------------
__global__ void __launch_bounds__(256)
softplus_kernel(float* __restrict__ z, const float* __restrict__ bias) {
    int idx = blockIdx.x * 256 + threadIdx.x;
    if (idx >= T_ * DM) return;
    float v = z[idx] + bias[idx % DM];
    z[idx] = (v > 20.0f) ? v : log1pf(__expf(v));
}

// ---------------- selective scan: one thread per (b, dm), 16 states in regs ----------------
__global__ void __launch_bounds__(128)
scan_kernel(const float* __restrict__ delta, const float* __restrict__ xc,
            const float* __restrict__ Bp, const float* __restrict__ Cp,
            const float* __restrict__ A_log, const float* __restrict__ Dv,
            float* __restrict__ y) {
    int idx = blockIdx.x * 128 + threadIdx.x;
    if (idx >= B_ * DM) return;
    int b = idx / DM;
    int d = idx % DM;
    float a[DS], h[DS];
#pragma unroll
    for (int s = 0; s < DS; ++s) {
        a[s] = -__expf(A_log[d * DS + s]);
        h[s] = 0.0f;
    }
    float Dd = Dv[d];
    for (int t = 0; t < L_; ++t) {
        int base = b * L_ + t;
        float dl  = delta[(size_t)base * DM + d];
        float xv  = xc[(size_t)base * DM + d];
        float dbu = dl * xv;
        const float* Bt = Bp + (size_t)base * DS;
        const float* Ct = Cp + (size_t)base * DS;
        float acc = 0.0f;
#pragma unroll
        for (int s = 0; s < DS; ++s) {
            h[s] = __expf(dl * a[s]) * h[s] + dbu * Bt[s];
            acc += h[s] * Ct[s];
        }
        y[(size_t)base * DM + d] = acc + Dd * xv;
    }
}

// ---------------- gate: yb = bf16( y * silu(res) ) ----------------
__global__ void __launch_bounds__(256)
gate_kernel(const float* __restrict__ y, const float* __restrict__ res,
            unsigned short* __restrict__ yb) {
    int idx = blockIdx.x * 256 + threadIdx.x;
    if (idx >= T_ * DM) return;
    float r = res[idx];
    float g = r * (1.0f / (1.0f + __expf(-r)));
    yb[idx] = f2bf(y[idx] * g);
}

static inline int grid1(int n) { return (n + 255) / 256; }
static inline int gridG(int tiles) { return (tiles + 7) / 8; }

extern "C" void kernel_launch(void* const* d_in, const int* in_sizes, int n_in,
                              void* d_out, int out_size, void* d_ws, size_t ws_size,
                              hipStream_t stream) {
    (void)in_sizes; (void)n_in; (void)out_size; (void)ws_size;
    const float* x_in   = (const float*)d_in[0];
    const float* W_in   = (const float*)d_in[1];
    const float* W_res  = (const float*)d_in[2];
    const float* W_out  = (const float*)d_in[3];
    const float* conv_w = (const float*)d_in[4];
    const float* conv_b = (const float*)d_in[5];
    const float* A_log  = (const float*)d_in[6];
    const float* Dv     = (const float*)d_in[7];
    const float* W_B    = (const float*)d_in[8];
    const float* W_C    = (const float*)d_in[9];
    const float* W_dt   = (const float*)d_in[10];
    const float* dt_w   = (const float*)d_in[11];
    const float* dt_b   = (const float*)d_in[12];

    char* ws = (char*)d_ws;
    size_t o = 0;
    auto take = [&](size_t bytes) -> void* {
        void* p = ws + o;
        o += (bytes + 255) & ~(size_t)255;
        return p;
    };
    unsigned short* Xb    = (unsigned short*)take((size_t)T_ * DIN * 2); // bf16 x_in
    unsigned short* WT    = (unsigned short*)take((size_t)DM * DIN * 2); // reused packed weight
    float*          xbuf  = (float*)take((size_t)T_ * DM * 4);           // x, later delta
    float*          resb  = (float*)take((size_t)T_ * DM * 4);           // res
    float*          xc32  = (float*)take((size_t)T_ * DM * 4);           // silu(conv) f32
    unsigned short* xc16  = (unsigned short*)take((size_t)T_ * DM * 2);  // bf16 xc, later bf16 gated y
    float*          ybuf  = (float*)take((size_t)T_ * DM * 4);           // scan output
    float*          Bpb   = (float*)take((size_t)T_ * DS * 4);
    float*          Cpb   = (float*)take((size_t)T_ * DS * 4);
    float*          t1    = (float*)take((size_t)T_ * DR * 4);
    unsigned short* t1b   = (unsigned short*)take((size_t)T_ * DR * 2);

    // x_in -> bf16
    cast_bf16_kernel<<<grid1(T_ * DIN), 256, 0, stream>>>(x_in, Xb, T_ * DIN);

    // x = x_in @ W_in   (M=2048, N=2048, K=1024)
    pack_transpose_kernel<<<grid1(DIN * DM), 256, 0, stream>>>(W_in, WT, DIN, DM);
    gemm_bf16_wmma_tiled<4><<<dim3(DM / 256, T_ / 64), 256, 0, stream>>>(Xb, WT, xbuf, nullptr, T_, DM, DIN);

    // res = x_in @ W_res
    pack_transpose_kernel<<<grid1(DIN * DM), 256, 0, stream>>>(W_res, WT, DIN, DM);
    gemm_bf16_wmma_tiled<4><<<dim3(DM / 256, T_ / 64), 256, 0, stream>>>(Xb, WT, resb, nullptr, T_, DM, DIN);

    // xc = silu(causal depthwise conv(x))
    conv_silu_kernel<<<grid1(T_ * DM), 256, 0, stream>>>(xbuf, conv_w, conv_b, xc32, xc16);

    // Bp = xc @ W_B   (N=16 -> simple kernel)
    pack_transpose_kernel<<<grid1(DM * DS), 256, 0, stream>>>(W_B, WT, DM, DS);
    gemm_bf16_wmma<<<gridG((T_ / 16) * (DS / 16)), 256, 0, stream>>>(xc16, WT, Bpb, nullptr, T_, DS, DM);

    // Cp = xc @ W_C
    pack_transpose_kernel<<<grid1(DM * DS), 256, 0, stream>>>(W_C, WT, DM, DS);
    gemm_bf16_wmma<<<gridG((T_ / 16) * (DS / 16)), 256, 0, stream>>>(xc16, WT, Cpb, nullptr, T_, DS, DM);

    // t1 = xc @ W_dt   (M=2048, N=128, K=2048; dual bf16 output)
    pack_transpose_kernel<<<grid1(DM * DR), 256, 0, stream>>>(W_dt, WT, DM, DR);
    gemm_bf16_wmma_tiled<2><<<dim3(DR / 128, T_ / 64), 256, 0, stream>>>(xc16, WT, t1, t1b, T_, DR, DM);

    // delta_pre = t1 @ dt_w   (M=2048, N=2048, K=128) -> xbuf (x dead)
    pack_transpose_kernel<<<grid1(DR * DM), 256, 0, stream>>>(dt_w, WT, DR, DM);
    gemm_bf16_wmma_tiled<4><<<dim3(DM / 256, T_ / 64), 256, 0, stream>>>(t1b, WT, xbuf, nullptr, T_, DM, DR);

    // delta = softplus(delta_pre + dt_b)
    softplus_kernel<<<grid1(T_ * DM), 256, 0, stream>>>(xbuf, dt_b);

    // selective scan -> ybuf (includes + D*xc)
    scan_kernel<<<(B_ * DM + 127) / 128, 128, 0, stream>>>(xbuf, xc32, Bpb, Cpb, A_log, Dv, ybuf);

    // gated y -> bf16 (reuse xc16)
    gate_kernel<<<grid1(T_ * DM), 256, 0, stream>>>(ybuf, resb, xc16);

    // out = y_gated @ W_out  (M=2048, N=1024, K=2048) -> d_out (f32)
    pack_transpose_kernel<<<grid1(DM * DIN), 256, 0, stream>>>(W_out, WT, DM, DIN);
    gemm_bf16_wmma_tiled<4><<<dim3(DIN / 256, T_ / 64), 256, 0, stream>>>(xc16, WT, (float*)d_out, nullptr, T_, DIN, DM);
}
